// GAT_8967891714352
// MI455X (gfx1250) — matrix operand order repaired and checked by reading, not compiled, not run on hardware
//
#include <hip/hip_runtime.h>
#include <hip/hip_bf16.h>
#include <math.h>

// Problem dimensions (fixed by the reference).
#define NN    6144      // nodes
#define FIN   512       // input features
#define NH    4         // heads
#define FHD   64        // per-head features
#define CO    40        // output classes
#define CP    48        // padded classes (3 x 16 tiles)
#define HF    (NH*FHD)  // 256 concat features
#define GAT_ALPHA 0.2f

typedef __bf16 bf16_t;
typedef __bf16 v16bf __attribute__((ext_vector_type(16)));
typedef float  v8f   __attribute__((ext_vector_type(8)));
typedef float  v2f   __attribute__((ext_vector_type(2)));

static __device__ inline v8f wmma_bf16(v16bf a, v16bf b, v8f c) {
  return __builtin_amdgcn_wmma_f32_16x16x32_bf16(false, a, false, b, (short)0, c, false, false);
}
static __device__ inline v8f wmma_f32k4(v2f a, v2f b, v8f c) {
  return __builtin_amdgcn_wmma_f32_16x16x4_f32(false, a, false, b, (short)0, c, false, false);
}

// ---------------------------------------------------------------------------
// Kernel 1: per-head feature transform  Wh[h] = x @ W[h]   (exact fp32 WMMA)
// One wave computes a 16-row x 64-col tile (4 accumulators), K=512 in steps of 4.
// Also emits WhT (bf16, [h][Fh][N]) so attention B-tiles are contiguous loads.
// ---------------------------------------------------------------------------
__global__ __launch_bounds__(32) void k_transform(
    const float* __restrict__ x, const float* __restrict__ W,
    float* __restrict__ Wh, bf16_t* __restrict__ WhT)
{
  const int i0   = blockIdx.x * 16;
  const int h    = blockIdx.y;
  const int lane = threadIdx.x & 31;
  const int half = lane >> 4;
  const int rl   = lane & 15;

  v8f acc[4];
  const v8f vz = {0.f,0.f,0.f,0.f,0.f,0.f,0.f,0.f};
  #pragma unroll
  for (int t = 0; t < 4; ++t) acc[t] = vz;

  const float* xr = x + (size_t)(i0 + rl) * FIN;
  const float* Wb = W + (size_t)h * FIN * FHD;

  for (int k = 0; k < FIN; k += 4) {
    // A (16x4 f32): lanes 0-15 hold K=k,k+1 ; lanes 16-31 hold K=k+2,k+3
    const float2 af = *(const float2*)(xr + k + 2 * half);
    v2f a; a.x = af.x; a.y = af.y;
    const float* wk = Wb + (size_t)(k + 2 * half) * FHD;
    #pragma unroll
    for (int t = 0; t < 4; ++t) {
      const int c = t * 16 + rl;
      v2f b; b.x = wk[c]; b.y = wk[FHD + c];
      acc[t] = wmma_f32k4(a, b, acc[t]);
    }
  }

  #pragma unroll
  for (int t = 0; t < 4; ++t) {
    const int c = t * 16 + rl;
    #pragma unroll
    for (int r = 0; r < 8; ++r) {
      const int row = i0 + r + 8 * half;     // C/D layout: lanes16-31 hold M=r+8
      const float v = acc[t][r];
      Wh [((size_t)h * NN + row) * FHD + c] = v;
      WhT[((size_t)h * FHD + c) * NN + row] = (bf16_t)v;
    }
  }
}

// ---------------------------------------------------------------------------
// Kernel 2: attention scores f1 = Wh@a1, f2 = Wh@a2 (one thread per (h,node))
// ---------------------------------------------------------------------------
__global__ void k_scores(const float* __restrict__ Wh,
                         const float* __restrict__ a1, const float* __restrict__ a2,
                         float* __restrict__ f1, float* __restrict__ f2)
{
  const int idx = blockIdx.x * blockDim.x + threadIdx.x;   // h*NN + i
  if (idx >= NH * NN) return;
  const int h = idx / NN;
  const float* row = Wh + (size_t)idx * FHD;
  const float* p1 = a1 + h * FHD;
  const float* p2 = a2 + h * FHD;
  float s1 = 0.f, s2 = 0.f;
  #pragma unroll 8
  for (int c = 0; c < FHD; ++c) { s1 += row[c] * p1[c]; s2 += row[c] * p2[c]; }
  f1[idx] = s1; f2[idx] = s2;
}

// ---------------------------------------------------------------------------
// Kernel 3/6: fused masked-softmax attention (flash style), bf16 WMMA P@Wh.
// One wave owns 16 output rows; streams 32-column adjacency tiles; online
// max/sum in registers; ELU epilogue. NT (compile-time) = 16-col output tiles,
// so all B loads and WMMAs fully unroll into clauses.
// ---------------------------------------------------------------------------
template <int NT>
__global__ __launch_bounds__(32) void k_attn(
    const float* __restrict__ adj,
    const float* __restrict__ f1, const float* __restrict__ f2,
    const bf16_t* __restrict__ whT, int headStrideT,
    float* __restrict__ out, int outPitch, int outColPerHead)
{
  const int i0   = blockIdx.x * 16;
  const int h    = blockIdx.y;
  const int lane = threadIdx.x & 31;
  const int half = lane >> 4;
  const int rl   = lane & 15;

  const float*  f1h  = f1  + (size_t)h * NN;
  const float*  f2h  = f2  + (size_t)h * NN;
  const float*  adjr = adj + (size_t)(i0 + rl) * NN;

  // Per-tile B base pointers (lane-fixed column, only j offset varies).
  const bf16_t* bbase[NT];
  #pragma unroll
  for (int t = 0; t < NT; ++t)
    bbase[t] = whT + (size_t)h * headStrideT + (size_t)(t * 16 + rl) * NN + 16 * half;

  const float f1r = f1h[i0 + rl];     // this lane's row score (dup in both halves)
  float m = -1e30f, l = 0.f;          // online softmax state for row rl

  v8f acc[NT];
  const v8f vz = {0.f,0.f,0.f,0.f,0.f,0.f,0.f,0.f};
  #pragma unroll
  for (int t = 0; t < NT; ++t) acc[t] = vz;

  for (int j0 = 0; j0 < NN; j0 += 32) {
    // A-layout column sets: lanes 0-15 -> K {0..15}, lanes 16-31 -> {16..31}
    const int jA = j0 + 16 * half;
    const int jB = jA + 8;
    const float4 a0 = *(const float4*)(adjr + jA);
    const float4 a1v = *(const float4*)(adjr + jA + 4);
    const float4 a2v = *(const float4*)(adjr + jB);
    const float4 a3v = *(const float4*)(adjr + jB + 4);
    const float4 g0 = *(const float4*)(f2h + jA);
    const float4 g1 = *(const float4*)(f2h + jA + 4);
    const float4 g2 = *(const float4*)(f2h + jB);
    const float4 g3 = *(const float4*)(f2h + jB + 4);

    float av[16] = {a0.x,a0.y,a0.z,a0.w, a1v.x,a1v.y,a1v.z,a1v.w,
                    a2v.x,a2v.y,a2v.z,a2v.w, a3v.x,a3v.y,a3v.z,a3v.w};
    float fv[16] = {g0.x,g0.y,g0.z,g0.w, g1.x,g1.y,g1.z,g1.w,
                    g2.x,g2.y,g2.z,g2.w, g3.x,g3.y,g3.z,g3.w};

    float ev[16];
    float mt = -1e30f;
    #pragma unroll
    for (int k = 0; k < 16; ++k) {
      float e = f1r + fv[k];
      e = (e > 0.f) ? e : GAT_ALPHA * e;          // LeakyReLU
      ev[k] = e;
      mt = fmaxf(mt, (av[k] > 0.f) ? e : -1e30f); // masked running tile max
    }
    mt = fmaxf(mt, __shfl_xor(mt, 16));           // combine the two K-halves of row rl

    const float mnew  = fmaxf(m, mt);
    const float scale = __expf(m - mnew);         // finite sentinel -> well-defined
    float p[16];
    float rs = 0.f;
    #pragma unroll
    for (int k = 0; k < 16; ++k) {
      const float pk = (av[k] > 0.f) ? __expf(ev[k] - mnew) : 0.f;  // explicit mask zero
      p[k] = pk; rs += pk;
    }
    rs += __shfl_xor(rs, 16);
    l = l * scale + rs;
    m = mnew;

    // Rescale accumulators: acc VGPR r belongs to row (r + 8*half).
    #pragma unroll
    for (int r = 0; r < 8; ++r) {
      const float sr = __shfl(scale, r + 8 * half);
      #pragma unroll
      for (int t = 0; t < NT; ++t) acc[t][r] *= sr;
    }

    // A operand (P tile) in bf16, contiguous k = 16*half + s, same order as B.
    v16bf pa;
    #pragma unroll
    for (int k = 0; k < 16; ++k) pa[k] = (bf16_t)p[k];

    // B tiles: contiguous 32B loads from WhT; fully unrolled -> load clause + WMMA chain.
    v16bf pb[NT];
    #pragma unroll
    for (int t = 0; t < NT; ++t) pb[t] = *(const v16bf*)(bbase[t] + j0);
    #pragma unroll
    for (int t = 0; t < NT; ++t) acc[t] = wmma_bf16(pa, pb[t], acc[t]);

    if (j0 + 32 < NN) __builtin_prefetch(adjr + j0 + 32, 0, 1);   // global_prefetch_b8
  }

  // Epilogue: normalize, ELU, scatter-store.
  const int colBase = h * outColPerHead;
  #pragma unroll
  for (int r = 0; r < 8; ++r) {
    const float lr  = __shfl(l, r + 8 * half);
    const int   row = i0 + r + 8 * half;
    #pragma unroll
    for (int t = 0; t < NT; ++t) {
      float v = acc[t][r] / lr;
      v = (v > 0.f) ? v : (__expf(v) - 1.f);       // ELU
      out[(size_t)row * outPitch + colBase + t * 16 + rl] = v;
    }
  }
}

// ---------------------------------------------------------------------------
// Kernel 4: output transform Wh2 = h @ Wo  (K=256, C padded 40->48, fp32 WMMA)
// ---------------------------------------------------------------------------
__global__ __launch_bounds__(32) void k_out_transform(
    const float* __restrict__ hbuf, const float* __restrict__ Wo,
    float* __restrict__ Wh2, bf16_t* __restrict__ Wh2T)
{
  const int i0   = blockIdx.x * 16;
  const int lane = threadIdx.x & 31;
  const int half = lane >> 4;
  const int rl   = lane & 15;

  v8f acc[3];
  const v8f vz = {0.f,0.f,0.f,0.f,0.f,0.f,0.f,0.f};
  #pragma unroll
  for (int t = 0; t < 3; ++t) acc[t] = vz;

  const float* xr = hbuf + (size_t)(i0 + rl) * HF;

  for (int k = 0; k < HF; k += 4) {
    const float2 af = *(const float2*)(xr + k + 2 * half);
    v2f a; a.x = af.x; a.y = af.y;
    const float* wk = Wo + (size_t)(k + 2 * half) * CO;
    #pragma unroll
    for (int t = 0; t < 3; ++t) {
      const int c = t * 16 + rl;
      v2f b;
      b.x = (c < CO) ? wk[c]      : 0.f;
      b.y = (c < CO) ? wk[CO + c] : 0.f;
      acc[t] = wmma_f32k4(a, b, acc[t]);
    }
  }

  #pragma unroll
  for (int t = 0; t < 3; ++t) {
    const int c = t * 16 + rl;
    #pragma unroll
    for (int r = 0; r < 8; ++r) {
      const int row = i0 + r + 8 * half;
      const float v = acc[t][r];
      Wh2 [(size_t)row * CP + c] = v;
      Wh2T[(size_t)c * NN + row] = (bf16_t)v;
    }
  }
}

// ---------------------------------------------------------------------------
// Kernel 5: output attention scores (dot over 40 valid columns).
// ---------------------------------------------------------------------------
__global__ void k_scores_o(const float* __restrict__ Wh2,
                           const float* __restrict__ ao1, const float* __restrict__ ao2,
                           float* __restrict__ f1o, float* __restrict__ f2o)
{
  const int i = blockIdx.x * blockDim.x + threadIdx.x;
  if (i >= NN) return;
  const float* row = Wh2 + (size_t)i * CP;
  float s1 = 0.f, s2 = 0.f;
  #pragma unroll 8
  for (int c = 0; c < CO; ++c) { s1 += row[c] * ao1[c]; s2 += row[c] * ao2[c]; }
  f1o[i] = s1; f2o[i] = s2;
}

// ---------------------------------------------------------------------------
// Kernel 7: row-wise log_softmax over the 40 valid columns.
// ---------------------------------------------------------------------------
__global__ void k_logsoftmax(const float* __restrict__ ob, float* __restrict__ out)
{
  const int i = blockIdx.x * blockDim.x + threadIdx.x;
  if (i >= NN) return;
  const float* r = ob + (size_t)i * CP;
  float mx = r[0];
  for (int c = 1; c < CO; ++c) mx = fmaxf(mx, r[c]);
  float s = 0.f;
  for (int c = 0; c < CO; ++c) s += __expf(r[c] - mx);
  const float ls = logf(s);
  for (int c = 0; c < CO; ++c) out[(size_t)i * CO + c] = r[c] - mx - ls;
}

// ---------------------------------------------------------------------------
extern "C" void kernel_launch(void* const* d_in, const int* in_sizes, int n_in,
                              void* d_out, int out_size, void* d_ws, size_t ws_size,
                              hipStream_t stream)
{
  (void)in_sizes; (void)n_in; (void)out_size; (void)ws_size;

  const float* x   = (const float*)d_in[0];
  const float* adj = (const float*)d_in[1];
  const float* W   = (const float*)d_in[2];
  const float* a1  = (const float*)d_in[3];
  const float* a2  = (const float*)d_in[4];
  const float* Wo  = (const float*)d_in[5];
  const float* ao1 = (const float*)d_in[6];
  const float* ao2 = (const float*)d_in[7];
  float* out = (float*)d_out;

  // Workspace carve-up (256B-aligned regions).
  char* ws = (char*)d_ws;
  size_t off = 0;
  auto carve = [&](size_t bytes) -> char* {
    char* p = ws + off;
    off += (bytes + 255) & ~(size_t)255;
    return p;
  };
  float*  Wh    = (float*) carve(sizeof(float)  * NH * NN * FHD);  // 6.3 MB
  bf16_t* WhT   = (bf16_t*)carve(sizeof(bf16_t) * NH * FHD * NN);  // 3.1 MB
  float*  f1    = (float*) carve(sizeof(float)  * NH * NN);
  float*  f2    = (float*) carve(sizeof(float)  * NH * NN);
  float*  hbuf  = (float*) carve(sizeof(float)  * NN * HF);        // 6.3 MB
  float*  Wh2   = (float*) carve(sizeof(float)  * NN * CP);
  bf16_t* Wh2T  = (bf16_t*)carve(sizeof(bf16_t) * CP * NN);
  float*  f1o   = (float*) carve(sizeof(float)  * NN);
  float*  f2o   = (float*) carve(sizeof(float)  * NN);
  float*  obuf  = (float*) carve(sizeof(float)  * NN * CP);

  const dim3 wave(32);

  // Layer 1: transform, scores, fused attention (per head).
  k_transform<<<dim3(NN / 16, NH), wave, 0, stream>>>(x, W, Wh, WhT);
  k_scores<<<(NH * NN + 255) / 256, 256, 0, stream>>>(Wh, a1, a2, f1, f2);
  k_attn<4><<<dim3(NN / 16, NH), wave, 0, stream>>>(adj, f1, f2, WhT,
                                                    /*headStrideT=*/FHD * NN,
                                                    hbuf, /*pitch=*/HF, /*colPerHead=*/FHD);

  // Layer 2: output transform, scores, fused attention, log_softmax.
  k_out_transform<<<dim3(NN / 16), wave, 0, stream>>>(hbuf, Wo, Wh2, Wh2T);
  k_scores_o<<<(NN + 255) / 256, 256, 0, stream>>>(Wh2, ao1, ao2, f1o, f2o);
  k_attn<3><<<dim3(NN / 16, 1), wave, 0, stream>>>(adj, f1o, f2o, Wh2T,
                                                   /*headStrideT=*/0,
                                                   obuf, /*pitch=*/CP, /*colPerHead=*/0);
  k_logsoftmax<<<(NN + 255) / 256, 256, 0, stream>>>(obuf, out);
}
